// EntropyCalculator_33457795236647
// MI455X (gfx1250) — matrix operand order repaired and checked by reading, not compiled, not run on hardware
//
#include <hip/hip_runtime.h>

// ---------------------------------------------------------------------------
// Types for CDNA5 WMMA (wave32): D(f32 16x16) = A(bf16 16x32) x B(bf16 32x16) + C
// ---------------------------------------------------------------------------
typedef __bf16 bf16;
typedef __attribute__((ext_vector_type(16))) bf16 bfrag;
typedef __attribute__((ext_vector_type(8)))  bf16 v8bf;
typedef __attribute__((ext_vector_type(8)))  float v8f;

constexpr int Bc = 256, Sc = 512, Dc = 128, Hc = 4, HDc = 32, FFc = 512, Vc = 256;
constexpr int Mc = Bc * Sc;  // 131072 rows

__device__ __forceinline__ int lane_id() { return threadIdx.x & 31; }

// A fragment: 16x32 bf16 tile, row-major source with leading dim `ld` (elements).
// Lane L holds row M=L&15; elems 0..7 = K (L>=16? 8:0)+0..7 ; elems 8..15 = K+16.
__device__ __forceinline__ bfrag load_frag_a(const bf16* p, int ld) {
  const int l = lane_id();
  const bf16* q = p + (size_t)(l & 15) * ld + ((l >> 4) << 3);
  v8bf lo = *(const v8bf*)q;
  v8bf hi = *(const v8bf*)(q + 16);
  bfrag r;
#pragma unroll
  for (int i = 0; i < 8; ++i) { r[i] = lo[i]; r[i + 8] = hi[i]; }
  return r;
}

// B fragment: 32x16 bf16 (KxN). Source indexed [n][k] contiguous in k, ld = row
// stride. Lane L holds col N=L&15; elems 0..7 = K (L>=16? 8:0)+0..7 ; elems 8..15 = K+16.
__device__ __forceinline__ bfrag load_frag_b(const bf16* p, int ld) {
  const int l = lane_id();
  const bf16* q = p + (size_t)(l & 15) * ld + ((l >> 4) << 3);
  v8bf lo = *(const v8bf*)q;
  v8bf hi = *(const v8bf*)(q + 16);
  bfrag r;
#pragma unroll
  for (int i = 0; i < 8; ++i) { r[i] = lo[i]; r[i + 8] = hi[i]; }
  return r;
}

__device__ __forceinline__ v8f wmma_bf16(bfrag a, bfrag b, v8f c) {
  return __builtin_amdgcn_wmma_f32_16x16x32_bf16(false, a, false, b, (short)0, c,
                                                 false, false);
}

// CDNA5 async global->LDS copy (ASYNCcnt-tracked, no VGPR round trip).
// lds_off: LDS byte address (low 32 bits of flat shared pointer).
__device__ __forceinline__ void async_b128(unsigned lds_off, const bf16* g) {
  asm volatile("global_load_async_to_lds_b128 %0, %1, off"
               :: "v"(lds_off), "v"(g) : "memory");
}
__device__ __forceinline__ void wait_async_le4() {
  asm volatile("s_wait_asynccnt 0x4" ::: "memory");
}
__device__ __forceinline__ void wait_async_0() {
  asm volatile("s_wait_asynccnt 0x0" ::: "memory");
}

// ---------------------------------------------------------------------------
// Generic GEMM: C[M,N] = A[M,K] @ W[N,K]^T + bias, optional relu.
// Tile 64x64, 256 threads (8 waves), each wave -> 16 rows x 32 cols (2 accums).
// K-step 64, double-buffered LDS filled by async global->LDS copies.
// ---------------------------------------------------------------------------
__global__ __launch_bounds__(256) void gemm_bf16(
    const bf16* __restrict__ A, const bf16* __restrict__ W,
    const float* __restrict__ bias, float* __restrict__ Cf,
    bf16* __restrict__ Cb, int N, int K, int relu) {
  __shared__ bf16 As[2][64][80];  // stride 80 elems = 160B (16B aligned rows)
  __shared__ bf16 Ws[2][64][80];
  const int n0 = blockIdx.x * 64, m0 = blockIdx.y * 64;
  const int t = threadIdx.x, wave = t >> 5, lane = t & 31;
  const int wr = (wave >> 1) * 16, wc = (wave & 1) * 32;
  // each thread owns 32B (16 bf16) of each 64x64 tile: row lr, cols lc..lc+15
  const int lr = t >> 2, lc = (t & 3) * 16;
  const bf16* agp = A + (size_t)(m0 + lr) * K + lc;
  const bf16* wgp = W + (size_t)(n0 + lr) * K + lc;
  unsigned aoff[2], woff[2];
  aoff[0] = (unsigned)(size_t)&As[0][lr][lc];
  aoff[1] = (unsigned)(size_t)&As[1][lr][lc];
  woff[0] = (unsigned)(size_t)&Ws[0][lr][lc];
  woff[1] = (unsigned)(size_t)&Ws[1][lr][lc];

  v8f c0 = {0, 0, 0, 0, 0, 0, 0, 0}, c1 = {0, 0, 0, 0, 0, 0, 0, 0};

  auto issue = [&](int buf, int k0) {
    async_b128(aoff[buf],      agp + k0);
    async_b128(aoff[buf] + 16, agp + k0 + 8);
    async_b128(woff[buf],      wgp + k0);
    async_b128(woff[buf] + 16, wgp + k0 + 8);
  };

  const int T = K >> 6;
  issue(0, 0);
  for (int tt = 0; tt < T; ++tt) {
    const int buf = tt & 1;
    const bool pre = (tt + 1 < T);
    if (pre) issue(buf ^ 1, (tt + 1) << 6);
    if (pre) wait_async_le4(); else wait_async_0();
    __syncthreads();  // all waves' async writes to buf are visible
    bfrag a0  = load_frag_a(&As[buf][wr][0], 80);
    bfrag a1  = load_frag_a(&As[buf][wr][32], 80);
    bfrag b00 = load_frag_b(&Ws[buf][wc][0], 80);
    bfrag b01 = load_frag_b(&Ws[buf][wc][32], 80);
    bfrag b10 = load_frag_b(&Ws[buf][wc + 16][0], 80);
    bfrag b11 = load_frag_b(&Ws[buf][wc + 16][32], 80);
    c0 = wmma_bf16(a0, b00, c0);
    c0 = wmma_bf16(a1, b01, c0);
    c1 = wmma_bf16(a0, b10, c1);
    c1 = wmma_bf16(a1, b11, c1);
    __syncthreads();  // protect buf from next iteration's async overwrite
  }

  const int rbase = m0 + wr + ((lane >> 4) << 3);
  const int ca = n0 + wc + (lane & 15);
  const float ba = bias ? bias[ca] : 0.f;
  const float bb = bias ? bias[ca + 16] : 0.f;
#pragma unroll
  for (int r = 0; r < 8; ++r) {
    float v0 = c0[r] + ba, v1 = c1[r] + bb;
    if (relu) { v0 = fmaxf(v0, 0.f); v1 = fmaxf(v1, 0.f); }
    const size_t i0 = (size_t)(rbase + r) * N + ca;
    if (Cf) { Cf[i0] = v0; Cf[i0 + 16] = v1; }
    if (Cb) { Cb[i0] = (bf16)v0; Cb[i0 + 16] = (bf16)v1; }
  }
}

// ---------------------------------------------------------------------------
// Repack qkv[M,3D] -> per-head q[BH,S,HD], k[BH,S,HD], vT[BH,HD,S]
// ---------------------------------------------------------------------------
__global__ __launch_bounds__(256) void repack_qkv(const bf16* __restrict__ qkv,
                                                  bf16* __restrict__ q,
                                                  bf16* __restrict__ k,
                                                  bf16* __restrict__ vt) {
  const size_t idx = (size_t)blockIdx.x * 256 + threadIdx.x;
  const int m = (int)(idx / (3 * Dc));
  const int c = (int)(idx % (3 * Dc));
  const int which = c >> 7, hd = c & 127, h = hd >> 5, d = hd & 31;
  const int b = m / Sc, s = m % Sc;
  const bf16 val = qkv[idx];
  const size_t bh = (size_t)b * Hc + h;
  if (which == 0)      q[(bh * Sc + s) * HDc + d] = val;
  else if (which == 1) k[(bh * Sc + s) * HDc + d] = val;
  else                 vt[(bh * HDc + d) * Sc + s] = val;
}

// ---------------------------------------------------------------------------
// Flash attention: one block = (b,h) x 128 q rows; each wave owns 16 q rows.
// Online softmax, probs staged through LDS (C-layout -> A-layout), AV via WMMA.
// ---------------------------------------------------------------------------
__global__ __launch_bounds__(256) void attn_flash(const bf16* __restrict__ qh,
                                                  const bf16* __restrict__ kh,
                                                  const bf16* __restrict__ vt,
                                                  bf16* __restrict__ obf) {
  __shared__ bf16 pbuf[8][16][32];
  const int bh = blockIdx.y;
  const int wave = threadIdx.x >> 5, lane = threadIdx.x & 31;
  const int q0 = blockIdx.x * 128 + wave * 16;
  const bf16* kbase = kh + (size_t)bh * Sc * HDc;
  const bf16* vbase = vt + (size_t)bh * HDc * Sc;
  bfrag qf = load_frag_a(qh + ((size_t)bh * Sc + q0) * HDc, HDc);
  float rm[8], rs[8];
  v8f o0 = {0, 0, 0, 0, 0, 0, 0, 0}, o1 = {0, 0, 0, 0, 0, 0, 0, 0};
#pragma unroll
  for (int r = 0; r < 8; ++r) { rm[r] = -1e30f; rs[r] = 0.f; }
  const float scale = 0.17677669529663689f;  // 1/sqrt(32)
  for (int kt = 0; kt < Sc / 16; kt += 2) {
#pragma unroll
    for (int sub = 0; sub < 2; ++sub) {
      const int key0 = (kt + sub) * 16;
      bfrag kf = load_frag_b(kbase + (size_t)key0 * HDc, HDc);
      v8f z = {0, 0, 0, 0, 0, 0, 0, 0};
      v8f sacc = wmma_bf16(qf, kf, z);
#pragma unroll
      for (int r = 0; r < 8; ++r) {
        float v = sacc[r] * scale;
        float mx = v;
#pragma unroll
        for (int off = 1; off < 16; off <<= 1)
          mx = fmaxf(mx, __shfl_xor(mx, off, 32));
        const float nm = fmaxf(rm[r], mx);
        const float corr = __expf(rm[r] - nm);
        const float p = __expf(v - nm);
        float ps = p;
#pragma unroll
        for (int off = 1; off < 16; off <<= 1) ps += __shfl_xor(ps, off, 32);
        rs[r] = rs[r] * corr + ps;
        rm[r] = nm;
        o0[r] *= corr;
        o1[r] *= corr;
        pbuf[wave][r + ((lane >> 4) << 3)][sub * 16 + (lane & 15)] = (bf16)p;
      }
    }
    __syncthreads();
    bfrag pf  = load_frag_a(&pbuf[wave][0][0], 32);
    bfrag vf0 = load_frag_b(vbase + (size_t)kt * 16, Sc);
    bfrag vf1 = load_frag_b(vbase + (size_t)16 * Sc + (size_t)kt * 16, Sc);
    o0 = wmma_bf16(pf, vf0, o0);
    o1 = wmma_bf16(pf, vf1, o1);
    __syncthreads();
  }
  const int b = bh / Hc, h = bh % Hc;
#pragma unroll
  for (int r = 0; r < 8; ++r) {
    const int row = q0 + r + ((lane >> 4) << 3);
    const float inv = 1.f / rs[r];
    const size_t base = ((size_t)b * Sc + row) * Dc + h * HDc + (lane & 15);
    obf[base] = (bf16)(o0[r] * inv);
    obf[base + 16] = (bf16)(o1[r] * inv);
  }
}

// ---------------------------------------------------------------------------
// LayerNorm helpers / kernels (one 128-wide row per 128-thread block)
// ---------------------------------------------------------------------------
__device__ __forceinline__ float row_sum_128(float v, float* red) {
#pragma unroll
  for (int off = 16; off > 0; off >>= 1) v += __shfl_xor(v, off, 32);
  if ((threadIdx.x & 31) == 0) red[threadIdx.x >> 5] = v;
  __syncthreads();
  const float s = red[0] + red[1] + red[2] + red[3];
  __syncthreads();
  return s;
}

__global__ __launch_bounds__(128) void embed_ln(const int* __restrict__ bytes,
                                                const float* __restrict__ emb,
                                                const float* __restrict__ pos,
                                                const float* __restrict__ g,
                                                const float* __restrict__ bt,
                                                float* __restrict__ hf,
                                                bf16* __restrict__ hb) {
  __shared__ float red[4];
  const size_t row = blockIdx.x;
  const int t = threadIdx.x;
  const int sp = (int)(row % Sc);
  const int tok = bytes[row];
  const float x = emb[(size_t)tok * Dc + t] + pos[(size_t)sp * Dc + t];
  const float mean = row_sum_128(x, red) * (1.f / Dc);
  const float dx = x - mean;
  const float var = row_sum_128(dx * dx, red) * (1.f / Dc);
  const float y = dx * rsqrtf(var + 1e-5f) * g[t] + bt[t];
  hf[row * Dc + t] = y;
  hb[row * Dc + t] = (bf16)y;
}

__global__ __launch_bounds__(128) void ln_residual(const float* __restrict__ base,
                                                   const float* __restrict__ delta,
                                                   const float* __restrict__ g,
                                                   const float* __restrict__ bt,
                                                   float* __restrict__ hf,
                                                   bf16* __restrict__ hb) {
  __shared__ float red[4];
  const size_t row = blockIdx.x;
  const int t = threadIdx.x;
  const float x = base[row * Dc + t] + delta[row * Dc + t];
  const float mean = row_sum_128(x, red) * (1.f / Dc);
  const float dx = x - mean;
  const float var = row_sum_128(dx * dx, red) * (1.f / Dc);
  const float y = dx * rsqrtf(var + 1e-5f) * g[t] + bt[t];
  hf[row * Dc + t] = y;
  hb[row * Dc + t] = (bf16)y;
}

// ---------------------------------------------------------------------------
// Entropy of softmax over V=256 logits, one row per 256-thread block.
// ---------------------------------------------------------------------------
__device__ __forceinline__ float red256(float v, float* red, bool is_max) {
#pragma unroll
  for (int off = 16; off > 0; off >>= 1) {
    const float o = __shfl_xor(v, off, 32);
    v = is_max ? fmaxf(v, o) : v + o;
  }
  if ((threadIdx.x & 31) == 0) red[threadIdx.x >> 5] = v;
  __syncthreads();
  float s = red[0];
#pragma unroll
  for (int w = 1; w < 8; ++w) s = is_max ? fmaxf(s, red[w]) : s + red[w];
  __syncthreads();
  return s;
}

__global__ __launch_bounds__(256) void entropy_rows(const float* __restrict__ logits,
                                                    float* __restrict__ ent) {
  __shared__ float red[8];
  const size_t row = blockIdx.x;
  const int t = threadIdx.x;
  const float x = logits[row * Vc + t];
  const float mx = red256(x, red, true);
  const float e = __expf(x - mx);
  const float Z = red256(e, red, false);
  const float p = e / Z;
  const float T = red256(p * logf(p + 1e-10f), red, false);
  if (t == 0) ent[row] = -T;
}

// avg over batch: deterministic fixed-tree reduction (no float atomics)
__global__ __launch_bounds__(256) void mean_over_batch(const float* __restrict__ ent,
                                                       float* __restrict__ avg) {
  __shared__ float red[8];
  const int s = blockIdx.x;
  const float v = ent[(size_t)threadIdx.x * Sc + s];
  const float t = red256(v, red, false);
  if (threadIdx.x == 0) avg[s] = t * (1.f / Bc);
}

// boundary + inclusive scan -> segment ids (written as float)
__global__ __launch_bounds__(512) void segment_scan(float* __restrict__ out) {
  __shared__ int sc[Sc];
  const int i = threadIdx.x;
  const float a = out[i];
  sc[i] = (i > 0 && a > 4.0f) ? 1 : 0;
  __syncthreads();
  for (int off = 1; off < Sc; off <<= 1) {
    const int v = sc[i];
    const int u = (i >= off) ? sc[i - off] : 0;
    __syncthreads();
    sc[i] = v + u;
    __syncthreads();
  }
  out[Sc + i] = (float)sc[i];
}

__global__ __launch_bounds__(256) void f32_to_bf16(const float* __restrict__ in,
                                                   bf16* __restrict__ out, int n) {
  const int i = blockIdx.x * 256 + threadIdx.x;
  if (i < n) out[i] = (bf16)in[i];
}

// ---------------------------------------------------------------------------
// Host side
// ---------------------------------------------------------------------------
extern "C" void kernel_launch(void* const* d_in, const int* in_sizes, int n_in,
                              void* d_out, int out_size, void* d_ws, size_t ws_size,
                              hipStream_t stream) {
  (void)in_sizes; (void)n_in; (void)out_size; (void)ws_size;
  const int*   in_bytes = (const int*)d_in[0];
  const float* emb_w  = (const float*)d_in[1];
  const float* pos_w  = (const float*)d_in[2];
  const float* ln_g   = (const float*)d_in[3];
  const float* ln_b   = (const float*)d_in[4];
  const float* attn_w = (const float*)d_in[5];
  const float* attn_b = (const float*)d_in[6];
  const float* out_w  = (const float*)d_in[7];
  const float* out_b  = (const float*)d_in[8];
  const float* ff1_w  = (const float*)d_in[9];
  const float* ff1_b  = (const float*)d_in[10];
  const float* ff2_w  = (const float*)d_in[11];
  const float* ff2_b  = (const float*)d_in[12];
  const float* n1_g   = (const float*)d_in[13];
  const float* n1_b   = (const float*)d_in[14];
  const float* n2_g   = (const float*)d_in[15];
  const float* n2_b   = (const float*)d_in[16];
  const float* proj_w = (const float*)d_in[17];
  const float* proj_b = (const float*)d_in[18];
  float* out = (float*)d_out;

  char* ws = (char*)d_ws;
  auto alloc = [&](size_t bytes) {
    char* p = ws;
    ws += (bytes + 255) & ~(size_t)255;
    return p;
  };
  float* h_f32 = (float*)alloc((size_t)Mc * Dc * 4);
  bf16*  h_bf  = (bf16*) alloc((size_t)Mc * Dc * 2);
  float* tmp   = (float*)alloc((size_t)Mc * Dc * 4);
  char*  act   = alloc((size_t)Mc * FFc * 2);  // qkv(bf16) / ffmid(bf16) / logits(f32)
  bf16*  qh  = (bf16*)alloc((size_t)Bc * Hc * Sc * HDc * 2);
  bf16*  kk  = (bf16*)alloc((size_t)Bc * Hc * Sc * HDc * 2);
  bf16*  vtb = (bf16*)alloc((size_t)Bc * Hc * HDc * Sc * 2);
  bf16*  obf = (bf16*)alloc((size_t)Mc * Dc * 2);
  float* ent = (float*)alloc((size_t)Mc * 4);
  bf16* attn_wb = (bf16*)alloc((size_t)2 * 3 * Dc * Dc * 2);
  bf16* out_wb  = (bf16*)alloc((size_t)2 * Dc * Dc * 2);
  bf16* ff1_wb  = (bf16*)alloc((size_t)2 * FFc * Dc * 2);
  bf16* ff2_wb  = (bf16*)alloc((size_t)2 * Dc * FFc * 2);
  bf16* proj_wb = (bf16*)alloc((size_t)Vc * Dc * 2);

  auto conv = [&](const float* src, bf16* dst, int n) {
    f32_to_bf16<<<(n + 255) / 256, 256, 0, stream>>>(src, dst, n);
  };
  conv(attn_w, attn_wb, 2 * 3 * Dc * Dc);
  conv(out_w,  out_wb,  2 * Dc * Dc);
  conv(ff1_w,  ff1_wb,  2 * FFc * Dc);
  conv(ff2_w,  ff2_wb,  2 * Dc * FFc);
  conv(proj_w, proj_wb, Vc * Dc);

  embed_ln<<<Mc, 128, 0, stream>>>(in_bytes, emb_w, pos_w, ln_g, ln_b, h_f32, h_bf);

  for (int l = 0; l < 2; ++l) {
    bf16* qkv = (bf16*)act;
    gemm_bf16<<<dim3(3 * Dc / 64, Mc / 64), 256, 0, stream>>>(
        h_bf, attn_wb + (size_t)l * 3 * Dc * Dc, attn_b + l * 3 * Dc,
        nullptr, qkv, 3 * Dc, Dc, 0);
    repack_qkv<<<(unsigned)((size_t)Mc * 3 * Dc / 256), 256, 0, stream>>>(qkv, qh, kk, vtb);
    attn_flash<<<dim3(Sc / 128, Bc * Hc), 256, 0, stream>>>(qh, kk, vtb, obf);
    gemm_bf16<<<dim3(Dc / 64, Mc / 64), 256, 0, stream>>>(
        obf, out_wb + (size_t)l * Dc * Dc, out_b + l * Dc, tmp, nullptr, Dc, Dc, 0);
    ln_residual<<<Mc, 128, 0, stream>>>(h_f32, tmp, n1_g + l * Dc, n1_b + l * Dc,
                                        h_f32, h_bf);
    bf16* mid = (bf16*)act;
    gemm_bf16<<<dim3(FFc / 64, Mc / 64), 256, 0, stream>>>(
        h_bf, ff1_wb + (size_t)l * FFc * Dc, ff1_b + l * FFc, nullptr, mid, FFc, Dc, 1);
    gemm_bf16<<<dim3(Dc / 64, Mc / 64), 256, 0, stream>>>(
        mid, ff2_wb + (size_t)l * Dc * FFc, ff2_b + l * Dc, tmp, nullptr, Dc, FFc, 0);
    ln_residual<<<Mc, 128, 0, stream>>>(h_f32, tmp, n2_g + l * Dc, n2_b + l * Dc,
                                        h_f32, h_bf);
  }

  float* logits = (float*)act;
  gemm_bf16<<<dim3(Vc / 64, Mc / 64), 256, 0, stream>>>(
      h_bf, proj_wb, proj_b, logits, nullptr, Vc, Dc, 0);
  entropy_rows<<<Mc, 256, 0, stream>>>(logits, ent);
  mean_over_batch<<<Sc, 256, 0, stream>>>(ent, out);
  segment_scan<<<1, Sc, 0, stream>>>(out);
}
